// SNN_VQVAE_29772713296279
// MI455X (gfx1250) — compile-verified, matrix operand order checked by reading, not run on hardware
//
#include <hip/hip_runtime.h>

// ---------------------------------------------------------------------------
// Types for CDNA5 WMMA + TDM (gfx1250, wave32)
// ---------------------------------------------------------------------------
typedef __attribute__((ext_vector_type(16))) __bf16 v16bf;
typedef __attribute__((ext_vector_type(8)))  float  v8f;
typedef unsigned int tdm_v4u __attribute__((ext_vector_type(4)));
typedef int          tdm_v8i __attribute__((ext_vector_type(8)));
typedef int          tdm_v4i __attribute__((ext_vector_type(4)));

__device__ __forceinline__ __bf16 f2bf(float f) {
  union { float f; unsigned u; } v; v.f = f;
  unsigned r = (v.u + 0x7FFFu + ((v.u >> 16) & 1u)) >> 16;   // RNE
  unsigned short s = (unsigned short)r;
  return __builtin_bit_cast(__bf16, s);
}
__device__ __forceinline__ __bf16 bfz() {
  return __builtin_bit_cast(__bf16, (unsigned short)0);
}

// ---------------------------------------------------------------------------
// Elementwise helpers
// ---------------------------------------------------------------------------
__global__ void k_f32_to_bf16(const float* __restrict__ in,
                              __bf16* __restrict__ out, long n) {
  long i = (long)blockIdx.x * blockDim.x + threadIdx.x;
  long stride = (long)gridDim.x * blockDim.x;
  for (; i < n; i += stride) out[i] = f2bf(in[i]);
}

__global__ void k_add_relu(const float* __restrict__ a,
                           const float* __restrict__ b,
                           float* __restrict__ out, long n) {
  long i = (long)blockIdx.x * blockDim.x + threadIdx.x;
  long stride = (long)gridDim.x * blockDim.x;
  for (; i < n; i += stride) {
    float v = a[i] + b[i];
    out[i] = v > 0.f ? v : 0.f;
  }
}

// LIF over T (tau=2, hard reset). X layout [T][per].
__global__ void k_lif(const float* __restrict__ X, __bf16* __restrict__ Sb,
                      float* __restrict__ Sf, long per, int T) {
  long i = (long)blockIdx.x * blockDim.x + threadIdx.x;
  if (i >= per) return;
  float v = 0.f;
  for (int t = 0; t < T; ++t) {
    float x = X[(long)t * per + i];
    v = (v + x) * 0.5f;
    float s = (v >= 1.f) ? 1.f : 0.f;
    v = v * (1.f - s);
    Sb[(long)t * per + i] = f2bf(s);
    if (Sf) Sf[(long)t * per + i] = s;
  }
}

// LIF with time-constant input (q broadcast over T).
__global__ void k_lif_const(const float* __restrict__ P,
                            float* __restrict__ Sf, __bf16* __restrict__ Sb,
                            long per, int T) {
  long i = (long)blockIdx.x * blockDim.x + threadIdx.x;
  if (i >= per) return;
  float x = P[i];
  float v = 0.f;
  for (int t = 0; t < T; ++t) {
    v = (v + x) * 0.5f;
    float s = (v >= 1.f) ? 1.f : 0.f;
    v = v * (1.f - s);
    Sf[(long)t * per + i] = s;
    Sb[(long)t * per + i] = f2bf(s);
  }
}

// xm = (1-alpha) * sum_t 0.8^(T-1-t) h[t] + alpha * mean_t h[t]
__global__ void k_vq_xm(const float* __restrict__ h,
                        const float* __restrict__ alphap,
                        float* __restrict__ xm) {
  int i = blockIdx.x * 256 + threadIdx.x;
  if (i >= 262144) return;
  float a = alphap[0];
  float c = 1.f;
  for (int k = 0; k < 15; ++k) c *= 0.8f;   // 0.8^(T-1)
  float sm = 0.f, mn = 0.f;
  for (int t = 0; t < 16; ++t) {
    float x = h[(long)t * 262144 + i];
    sm += c * x; mn += x; c *= 1.25f;
  }
  xm[i] = (1.f - a) * sm + a * mn * (1.f / 16.f);
}

// xm [b][c][p] -> Xd bf16 [c][b*256+p] (GEMM B operand), flat f32 [n][c]
__global__ void k_xm_prep(const float* __restrict__ xm,
                          __bf16* __restrict__ Xd, float* __restrict__ flat) {
  int i = blockIdx.x * 256 + threadIdx.x;
  if (i >= 262144) return;
  int p = i & 255, c = (i >> 8) & 255, b = i >> 16;
  float v = xm[i];
  int n = b * 256 + p;
  Xd[c * 1024 + n] = f2bf(v);
  flat[n * 256 + c] = v;
}

__global__ void k_enorm(const float* __restrict__ emb, float* __restrict__ en) {
  int e = blockIdx.x * 256 + threadIdx.x;
  if (e >= 1024) return;
  float s = 0.f;
  for (int c = 0; c < 256; ++c) { float v = emb[e * 256 + c]; s += v * v; }
  en[e] = s;
}

// Per VQ position: argmin_e (|e|^2 - 2 dot[e][n]), gather qraw, accumulate SSE.
__global__ void k_vq_argmin(const float* __restrict__ dot,
                            const float* __restrict__ en,
                            const float* __restrict__ flat,
                            const float* __restrict__ emb,
                            float* __restrict__ q, float* __restrict__ acc) {
  int n = blockIdx.x;                 // 0..1023
  __shared__ float sv[256];
  __shared__ int   si[256];
  float best = 3.0e38f; int bi = 0;
  for (int e = threadIdx.x; e < 1024; e += 256) {
    float d = en[e] - 2.f * dot[(long)e * 1024 + n];
    if (d < best) { best = d; bi = e; }
  }
  sv[threadIdx.x] = best; si[threadIdx.x] = bi;
  __syncthreads();
  for (int s = 128; s > 0; s >>= 1) {
    if (threadIdx.x < s && sv[threadIdx.x + s] < sv[threadIdx.x]) {
      sv[threadIdx.x] = sv[threadIdx.x + s];
      si[threadIdx.x] = si[threadIdx.x + s];
    }
    __syncthreads();
  }
  int e = si[0];
  __syncthreads();
  int b = n >> 8, p = n & 255;
  int c = threadIdx.x;                // 256 channels, one each
  float qv = emb[e * 256 + c];
  float xv = flat[n * 256 + c];
  q[((long)(b * 256 + c)) * 256 + p] = qv;
  float dd = qv - xv;
  sv[threadIdx.x] = dd * dd;
  __syncthreads();
  for (int s = 128; s > 0; s >>= 1) {
    if (threadIdx.x < s) sv[threadIdx.x] += sv[threadIdx.x + s];
    __syncthreads();
  }
  if (threadIdx.x == 0) atomicAdd(acc, sv[0]);
}

// PSP (tau_s = 2) of two sequences + accumulated squared error over all t.
__global__ void k_psp_loss(const float* __restrict__ q,
                           const float* __restrict__ h,
                           long per, int T, float* __restrict__ acc) {
  long i = (long)blockIdx.x * 256 + threadIdx.x;
  float sum = 0.f;
  if (i < per) {
    float sq = 0.f, sh = 0.f;
    for (int t = 0; t < T; ++t) {
      sq = (sq + q[(long)t * per + i]) * 0.5f;
      sh = (sh + h[(long)t * per + i]) * 0.5f;
      float d = sq - sh;
      sum += d * d;
    }
  }
  __shared__ float red[256];
  red[threadIdx.x] = sum;
  __syncthreads();
  for (int s = 128; s > 0; s >>= 1) {
    if (threadIdx.x < s) red[threadIdx.x] += red[threadIdx.x + s];
    __syncthreads();
  }
  if (threadIdx.x == 0) atomicAdd(acc, red[0]);
}

__global__ void k_zero2(float* a) { if (threadIdx.x < 2) a[threadIdx.x] = 0.f; }

__global__ void k_loss_final(const float* __restrict__ acc,
                             float* __restrict__ out) {
  if (blockIdx.x == 0 && threadIdx.x == 0)
    out[16384] = 1.25f * (acc[0] / 262144.f + acc[1] / 4194304.f);
}

// Small transposed 3x3 stride-2 convs in decoder (scalar path; ~2.4 GFLOP).
__global__ void k_convt3(const float* __restrict__ X,
                         const float* __restrict__ Wt,
                         const float* __restrict__ bias,
                         float* __restrict__ Y,
                         int Cin, int Cout, int Hi, int Wi, int Ho, int Wo,
                         int nimg) {
  long idx = (long)blockIdx.x * 256 + threadIdx.x;
  long tot = (long)nimg * Cout * Ho * Wo;
  if (idx >= tot) return;
  int p = (int)(idx % (Ho * Wo));
  long tmp = idx / (Ho * Wo);
  int o = (int)(tmp % Cout);
  int img = (int)(tmp / Cout);
  int oh = p / Wo, ow = p % Wo;
  float accv = bias[o];
  for (int kh = 0; kh < 3; ++kh) {
    int th = oh + 1 - kh;
    if (th < 0 || (th & 1)) continue;
    int ih = th >> 1;
    if (ih >= Hi) continue;
    for (int kw = 0; kw < 3; ++kw) {
      int tw = ow + 1 - kw;
      if (tw < 0 || (tw & 1)) continue;
      int iw = tw >> 1;
      if (iw >= Wi) continue;
      for (int ci = 0; ci < Cin; ++ci) {
        accv += X[(((long)img * Cin + ci) * Hi + ih) * Wi + iw] *
                Wt[((long)ci * Cout + o) * 9 + kh * 3 + kw];
      }
    }
  }
  Y[idx] = accv;
}

// recon = sum_t 0.8^(T-1-t) * y5[t]
__global__ void k_memout(const float* __restrict__ y, float* __restrict__ out) {
  int i = blockIdx.x * 256 + threadIdx.x;
  if (i >= 16384) return;
  int b = i >> 12, p = i & 4095;
  float c = 1.f;
  for (int k = 0; k < 15; ++k) c *= 0.8f;
  float s = 0.f;
  for (int t = 0; t < 16; ++t) {
    s += c * y[(((long)(t * 4 + b)) << 12) + p];
    c *= 1.25f;
  }
  out[i] = s;
}

// ---------------------------------------------------------------------------
// Implicit-GEMM convolution via V_WMMA_F32_16X16X32_BF16.
//   Y[img][co][oh][ow] = g[co] * (sum_{ci,kh,kw} W * X) + b[co]
// Tile: 64 (Cout) x 128 (pixels), K-step 32, 256 threads = 8 wave32 waves,
// 4 WMMA accumulators per wave. Weight slab (all filter taps of a 32-channel
// Cin slice) is DMA'd into LDS by the Tensor Data Mover (tensor_load_to_lds,
// TENSORcnt-tracked) and waited on before the barrier; the im2col gather for the B tile follows.
// wtrans=1 indexes W as [Cin][Cout] (ConvT 1x1) and uses the VALU fallback.
// ---------------------------------------------------------------------------
#define TM 64
#define TN 128
#define TK 32
#define MAXK2 9

__global__ __launch_bounds__(256)
void k_conv_wmma(const __bf16* __restrict__ X, const __bf16* __restrict__ W,
                 float* __restrict__ Y,
                 const float* __restrict__ gamma, const float* __restrict__ beta,
                 int Cin, int Cout, int H, int Wd, int Ho, int Wo,
                 int ksize, int pad, int stride, int wtrans) {
  __shared__ __bf16 sW[TM * TK * MAXK2];   // [co][ci*K2 + r], row stride TK*K2
  __shared__ __bf16 sB[TK][TN + 8];

  const int img = blockIdx.z;
  const int m0 = blockIdx.y * TM;
  const int n0 = blockIdx.x * TN;
  const int tid = threadIdx.x;
  const int lane = tid & 31;
  const int wv = tid >> 5;           // wave 0..7
  const int mi = wv & 3;             // M subtile
  const int nj = (wv >> 2) * 4;      // first of four N subtiles
  const int K2 = ksize * ksize;
  const int HWo = Ho * Wo;
  const int wrow = TK * K2;          // LDS weight row stride

  const __bf16* Ximg = X + (long)img * Cin * H * Wd;
  v8f acc[4] = {v8f{}, v8f{}, v8f{}, v8f{}};

  for (int ci0 = 0; ci0 < Cin; ci0 += TK) {
    // ---- stage weight slab sW[co][ci*K2+r] for the whole Cin slice --------
    if (!wtrans) {
      if (tid < 32) {
        // Tensor DMA: 2D tile, x = ci*K2+r contiguous, y = co (stride Cin*K2)
        unsigned lds_off = (unsigned)(unsigned long long)(&sW[0]);
        unsigned long long ga =
            (unsigned long long)(const void*)(W + ((long)m0 * Cin + ci0) * K2);
        long td0 = (long)(Cin - ci0) * K2;     // remaining x extent (OOB -> 0)
        int  td1 = Cout - m0;                  // remaining rows
        long st0 = (long)Cin * K2;             // row stride, data_size units
        tdm_v4u g0;
        g0[0] = 1u;                                     // count=1, user D#
        g0[1] = lds_off;                                // lds_addr
        g0[2] = (unsigned)(ga & 0xFFFFFFFFu);           // global_addr[31:0]
        g0[3] = (unsigned)((ga >> 32) & 0x01FFFFFFu) |  // global_addr[56:32]
                (2u << 30);                             // type=2 (image)
        tdm_v8i g1;
        g1[0] = 1 << 16;                                // data_size=1 (2 bytes)
        g1[1] = (int)((td0 & 0xFFFF) << 16);            // dim0[15:0] @63:48
        g1[2] = (int)(((td0 >> 16) & 0xFFFF) |          // dim0[31:16]
                      ((unsigned)(td1 & 0xFFFF) << 16));// dim1[15:0]
        g1[3] = (int)(((td1 >> 16) & 0xFFFF) |          // dim1[31:16]
                      ((unsigned)wrow << 16));          // tile_dim0
        g1[4] = TM & 0xFFFF;                            // tile_dim1, dim2=0
        g1[5] = (int)(st0 & 0xFFFFFFFF);                // dim0_stride[31:0]
        g1[6] = (int)((st0 >> 32) & 0xFFFF);            // dim0_stride[47:32]
        g1[7] = 0;
        tdm_v4i gz = {0, 0, 0, 0};
        tdm_v8i gz8 = {0, 0, 0, 0, 0, 0, 0, 0};
        __builtin_amdgcn_tensor_load_to_lds(g0, g1, gz, gz, gz8, 0);
        __builtin_amdgcn_s_wait_tensorcnt(0);
      }
    } else {
      for (int i = tid; i < TM * wrow; i += 256) {
        int co = i / wrow, rest = i % wrow;
        int ci = rest / K2, r = rest % K2;
        int gco = m0 + co, gci = ci0 + ci;
        __bf16 v = bfz();
        if (gco < Cout && gci < Cin)
          v = W[((long)gci * Cout + gco) * K2 + r];
        sW[co * wrow + rest] = v;
      }
    }
    if (ci0 + TK < Cin) {
      // hint L2: next Cin slab of activations (global_prefetch_b8)
      __builtin_prefetch(Ximg + (long)(ci0 + TK + lane) * H * Wd, 0, 1);
    }
    __syncthreads();

    for (int r = 0; r < K2; ++r) {
      const int kh = r / ksize, kw = r % ksize;
      // stage shifted-input tile 32x128 (implicit im2col)
      for (int i = tid; i < TK * TN; i += 256) {
        int ci = i / TN, n = i % TN;
        int gn = n0 + n, gci = ci0 + ci;
        __bf16 v = bfz();
        if (gn < HWo && gci < Cin) {
          int oh = gn / Wo, ow = gn % Wo;
          int ih = oh * stride + kh - pad;
          int iw = ow * stride + kw - pad;
          if (ih >= 0 && ih < H && iw >= 0 && iw < Wd)
            v = Ximg[((long)gci * H + ih) * Wd + iw];
        }
        sB[ci][n] = v;
      }
      __syncthreads();

      // Pack bf16 fragments per ISA 7.12.2 (16-bit A 16x32 / B 32x16):
      // lane L: row/col = L%16; K offset 8 for lanes 16-31; +16 for elems 8-15.
      const int row = lane & 15;
      const int khi = (lane >> 4) << 3;
      v16bf a;
#pragma unroll
      for (int e = 0; e < 16; ++e) {
        int k = ((e >> 3) << 4) + khi + (e & 7);
        a[e] = sW[(mi * 16 + row) * wrow + k * K2 + r];
      }
#pragma unroll
      for (int j = 0; j < 4; ++j) {
        v16bf b;
#pragma unroll
        for (int e = 0; e < 16; ++e) {
          int k = ((e >> 3) << 4) + khi + (e & 7);
          b[e] = sB[k][(nj + j) * 16 + row];
        }
        acc[j] = __builtin_amdgcn_wmma_f32_16x16x32_bf16(
            false, a, false, b, (short)0, acc[j], false, false);
      }
      __syncthreads();
    }
  }

  // D layout: lane L col = L%16; VGPR v row = v + 8*(L/16)
  float* Yimg = Y + (long)img * Cout * HWo;
  const int col = lane & 15;
  const int rbase = (lane >> 4) << 3;
#pragma unroll
  for (int v = 0; v < 8; ++v) {
    int gm = m0 + mi * 16 + rbase + v;
    if (gm >= Cout) continue;
    float sc = gamma ? gamma[gm] : 1.f;
    float bi = beta ? beta[gm] : 0.f;
#pragma unroll
    for (int j = 0; j < 4; ++j) {
      int gn = n0 + (nj + j) * 16 + col;
      if (gn < HWo) Yimg[(long)gm * HWo + gn] = acc[j][v] * sc + bi;
    }
  }
}

// ---------------------------------------------------------------------------
// Orchestration
// ---------------------------------------------------------------------------
extern "C" void kernel_launch(void* const* d_in, const int* in_sizes, int n_in,
                              void* d_out, int out_size, void* d_ws, size_t ws_size,
                              hipStream_t stream) {
  (void)in_sizes; (void)n_in; (void)out_size; (void)ws_size;

  // Input index map (setup_inputs dict order, flattened)
  enum {
    IN_X = 0, IN_BASE_W = 1, IN_BASE_G = 2, IN_BASE_B = 3,
    IN_L1B1 = 4, IN_L1B2 = 10, IN_L2B1 = 16, IN_L2B2 = 25,
    IN_L3B1 = 31, IN_L3B2 = 40,
    IN_ALPHA = 46, IN_EMB = 47, IN_POIS_W = 48, IN_POIS_G = 49, IN_POIS_B = 50,
    IN_DW2 = 51, IN_DB2 = 52, IN_DW3 = 53, IN_DB3 = 54,
    IN_DW4 = 55, IN_DB4 = 56, IN_DW5 = 57, IN_DB5 = 58
  };
  auto f = [&](int i) { return (const float*)d_in[i]; };

  // Workspace partition
  char* ws = (char*)d_ws;
  size_t off = 0;
  auto take = [&](size_t bytes) {
    size_t o = off; off += (bytes + 255) & ~(size_t)255; return o;
  };
  float*  F0    = (float*) (ws + take(67108864));   // 64img x 64ch x 64^2 f32
  float*  F1    = (float*) (ws + take(67108864));
  float*  F2    = (float*) (ws + take(67108864));
  __bf16* H0b   = (__bf16*)(ws + take(33554432));
  __bf16* H1b   = (__bf16*)(ws + take(33554432));
  __bf16* WBb   = (__bf16*)(ws + take(2097152));    // weight bf16 scratch
  float*  HSPK  = (float*) (ws + take(16777216));   // h  [16][4*256*16*16]
  float*  QSPK  = (float*) (ws + take(16777216));   // qspk f32
  __bf16* QSPKB = (__bf16*)(ws + take(8388608));    // qspk bf16
  float*  XM    = (float*) (ws + take(1048576));
  __bf16* XD    = (__bf16*)(ws + take(524288));     // [256][1024]
  float*  FLAT  = (float*) (ws + take(1048576));
  float*  EMBN  = (float*) (ws + take(65536));
  float*  DOT   = (float*) (ws + take(4194304));    // [1024][1024]
  float*  Qf    = (float*) (ws + take(1048576));
  __bf16* QB    = (__bf16*)(ws + take(524288));
  float*  Pois  = (float*) (ws + take(1048576));
  float*  ACC   = (float*) (ws + take(256));

  auto cvt = [&](const float* src, __bf16* dst, long n) {
    unsigned g = (unsigned)((n + 255) / 256);
    k_f32_to_bf16<<<g, 256, 0, stream>>>(src, dst, n);
  };
  auto conv = [&](const __bf16* Xb, const float* Wf, long wN, float* Yf,
                  const float* g, const float* bta, int Cin, int Cout,
                  int H, int W, int ks, int pad, int strd, int wtr, int nimg) {
    cvt(Wf, WBb, wN);
    int Ho = (H + 2 * pad - ks) / strd + 1;
    int Wo = (W + 2 * pad - ks) / strd + 1;
    dim3 g3((unsigned)((Ho * Wo + TN - 1) / TN),
            (unsigned)((Cout + TM - 1) / TM), (unsigned)nimg);
    k_conv_wmma<<<g3, 256, 0, stream>>>(Xb, WBb, Yf, g, bta, Cin, Cout,
                                        H, W, Ho, Wo, ks, pad, strd, wtr);
  };
  auto lif = [&](const float* X, __bf16* Sb, float* Sf, long per) {
    unsigned g = (unsigned)((per + 255) / 256);
    k_lif<<<g, 256, 0, stream>>>(X, Sb, Sf, per, 16);
  };
  auto addrelu = [&](const float* a, const float* b, float* o, long n) {
    unsigned g = (unsigned)((n + 255) / 256);
    k_add_relu<<<g, 256, 0, stream>>>(a, b, o, n);
  };

  k_zero2<<<1, 32, 0, stream>>>(ACC);

  // ---- Encoder base: conv(1->64) + BN + LIF -------------------------------
  cvt(f(IN_X), H0b, 262144L * 4);                    // x -> bf16 (64 imgs)
  conv(H0b, f(IN_BASE_W), 576, F0, f(IN_BASE_G), f(IN_BASE_B),
       1, 64, 64, 64, 3, 1, 1, 0, 64);
  lif(F0, H0b, F1, 4L * 64 * 64 * 64);               // spikes bf16 + f32

  // ---- Residual blocks -----------------------------------------------------
  // state: F1 = block input f32, H0b = block input bf16
  auto resblock = [&](int base, int cin, int cout, int H, int s, bool down,
                      float* outF) {
    int Ho = (H + 2 - 3) / s + 1;
    long nout = 64L * cout * Ho * Ho;
    conv(H0b, f(base + 0), (long)cout * cin * 9, F0, f(base + 1), f(base + 2),
         cin, cout, H, H, 3, 1, s, 0, 64);
    lif(F0, H1b, nullptr, nout / 16);
    conv(H1b, f(base + 3), (long)cout * cout * 9, F2, f(base + 4), f(base + 5),
         cout, cout, Ho, Ho, 3, 1, 1, 0, 64);
    const float* sc = F1;
    if (down) {
      conv(H0b, f(base + 6), (long)cout * cin, F0, f(base + 7), f(base + 8),
           cin, cout, H, H, 1, 0, s, 0, 64);
      sc = F0;
    }
    addrelu(sc, F2, outF, nout);
  };

  resblock(IN_L1B1, 64, 64, 64, 1, false, F1);  cvt(F1, H0b, 64L * 64 * 4096);
  resblock(IN_L1B2, 64, 64, 64, 1, false, F1);  cvt(F1, H0b, 64L * 64 * 4096);
  resblock(IN_L2B1, 64, 128, 64, 2, true, F1);  cvt(F1, H0b, 64L * 128 * 1024);
  resblock(IN_L2B2, 128, 128, 32, 1, false, F1); cvt(F1, H0b, 64L * 128 * 1024);
  resblock(IN_L3B1, 128, 256, 32, 2, true, F1); cvt(F1, H0b, 64L * 256 * 256);
  resblock(IN_L3B2, 256, 256, 16, 1, false, HSPK);   // h f32 [16][262144]

  // ---- Vector quantizer ----------------------------------------------------
  k_vq_xm<<<1024, 256, 0, stream>>>(HSPK, f(IN_ALPHA), XM);
  k_xm_prep<<<1024, 256, 0, stream>>>(XM, XD, FLAT);
  k_enorm<<<4, 256, 0, stream>>>(f(IN_EMB), EMBN);
  // dot[e][n] = emb . xm  as a 1024x1024x256 WMMA GEMM
  conv(XD, f(IN_EMB), 262144, DOT, nullptr, nullptr,
       256, 1024, 1, 1024, 1, 0, 1, 0, 1);
  k_vq_argmin<<<1024, 256, 0, stream>>>(DOT, EMBN, FLAT, f(IN_EMB), Qf, ACC);

  // pois conv(q) + BN, then LIF over broadcast-T
  cvt(Qf, QB, 262144);
  conv(QB, f(IN_POIS_W), 65536, Pois, f(IN_POIS_G), f(IN_POIS_B),
       256, 256, 16, 16, 1, 0, 1, 0, 4);
  k_lif_const<<<1024, 256, 0, stream>>>(Pois, QSPK, QSPKB, 262144, 16);

  // PSP MSE between qspk and h
  k_psp_loss<<<1024, 256, 0, stream>>>(QSPK, HSPK, 262144, 16, ACC + 1);

  // ---- Decoder -------------------------------------------------------------
  // ConvT 1x1 (256->128) and (128->64) as transposed-weight WMMA GEMMs
  conv(QSPKB, f(IN_DW2), 32768, F0, nullptr, f(IN_DB2),
       256, 128, 16, 16, 1, 0, 1, 1, 64);
  cvt(F0, H0b, 64L * 128 * 256);
  conv(H0b, f(IN_DW3), 8192, F1, nullptr, f(IN_DB3),
       128, 64, 16, 16, 1, 0, 1, 1, 64);
  // ConvT 3x3 stride-2 (64->32, 32->1): scalar
  {
    long tot4 = 64L * 32 * 32 * 32;
    k_convt3<<<(unsigned)((tot4 + 255) / 256), 256, 0, stream>>>(
        F1, f(IN_DW4), f(IN_DB4), F0, 64, 32, 16, 16, 32, 32, 64);
    long tot5 = 64L * 1 * 64 * 64;
    k_convt3<<<(unsigned)((tot5 + 255) / 256), 256, 0, stream>>>(
        F0, f(IN_DW5), f(IN_DB5), F2, 32, 1, 32, 32, 64, 64, 64);
  }
  // recon = memout(y5) -> d_out[0:16384]; loss -> d_out[16384]
  k_memout<<<64, 256, 0, stream>>>(F2, (float*)d_out);
  k_loss_final<<<1, 32, 0, stream>>>(ACC, (float*)d_out);
}
